// myLSTM_6433861009488
// MI455X (gfx1250) — matrix operand order repaired and checked by reading, not compiled, not run on hardware
//
#include <hip/hip_runtime.h>

// Problem dims (fixed by reference)
#define B_  128
#define T_  512
#define D_  256
#define H_  512
#define R_  64
#define G4H (4*H_)

typedef __attribute__((ext_vector_type(16))) __bf16        v16bf;
typedef __attribute__((ext_vector_type(8)))  float         v8f;
typedef __attribute__((ext_vector_type(4)))  unsigned int  u32x4;
typedef __attribute__((ext_vector_type(4)))  float         f32x4;

union AFrag {                       // 16x32 bf16 A fragment (8 VGPRs / lane)
  v16bf          v;
  u32x4          q[2];
  unsigned short h[16];
};

// Hardware f32->bf16 (RNE) — lowers to v_cvt_pk_bf16_f32.
__device__ __forceinline__ unsigned short f2bf(float f) {
  union { __bf16 b; unsigned short u; } cv;
  cv.b = (__bf16)f;
  return cv.u;
}
__device__ __forceinline__ float fsig(float x)  { return 1.0f / (1.0f + __expf(-x)); }
__device__ __forceinline__ float ftanh(float x) { return 1.0f - 2.0f / (__expf(2.0f * x) + 1.0f); }

__device__ __forceinline__ v8f wmma_bf16(v16bf a, v16bf b, v8f c) {
  // D = A(16x32 bf16) * B(32x16 bf16) + C(16x16 f32)
  return __builtin_amdgcn_wmma_f32_16x16x32_bf16(false, a, false, b, (short)0, c,
                                                 false, false);
}

// ---------------------------------------------------------------------------
// Pre-pack an fp32 weight matrix [KTsrc*32 x Ntiles*16] into bf16 WMMA
// B-fragments. Fragment storage: frag (nt*KTtot + ktOff + kt) = 32 lanes x
// 16 halves contiguous (1 KB), so the hot loop reads 32 B/lane contiguous.
// B-frag per-lane layout: n = lane&15, k = (i&8)*2 + (lane>>4)*8 + (i&7)  (i = half idx).
// ---------------------------------------------------------------------------
__global__ __launch_bounds__(256) void pack_frags_kernel(
    const float* __restrict__ src, int ldn,
    int KTsrc, int Ntiles, int KTtot, int ktOff,
    unsigned short* __restrict__ dst) {
  int gid = blockIdx.x * blockDim.x + threadIdx.x;
  int total = KTsrc * Ntiles * 32;
  if (gid >= total) return;
  int frag = gid >> 5, lane = gid & 31;
  int nt = frag / KTsrc, kt = frag % KTsrc;
  int sub = lane & 15, khi = lane >> 4;
  unsigned short* out = dst + ((size_t)(nt * KTtot + ktOff + kt)) * 512
                            + (size_t)lane * 16;
  int col = nt * 16 + sub;
#pragma unroll
  for (int i = 0; i < 16; ++i) {
    int row = kt * 32 + (i & 8) * 2 + khi * 8 + (i & 7);
    out[i] = f2bf(src[(size_t)row * ldn + col]);
  }
}

// ---------------------------------------------------------------------------
// xr = bf16( X[B,T,Din] @ W[Din,64] ), stored t-major: xr[t][b][64] bf16 so
// each scan step reads one contiguous 2 KB block per workgroup.
// One wave per 16-row (t) tile. Per K-step: clause of 4 B-frag loads into
// distinct regs, then 4 independent WMMAs.
// ---------------------------------------------------------------------------
__global__ __launch_bounds__(256) void gemm_xr_kernel(
    const float* __restrict__ X, int Din, int KT,
    const unsigned short* __restrict__ Wfr,
    unsigned short* __restrict__ xr) {
  int gw = (blockIdx.x * blockDim.x + threadIdx.x) >> 5;
  int lane = threadIdx.x & 31;
  const int TT = T_ / 16;
  int b = gw / TT;
  int t0 = (gw % TT) * 16;
  if (b >= B_) return;
  int sub = lane & 15, khi = lane >> 4;
  const float* xrow = X + ((size_t)b * T_ + t0 + sub) * Din;

  v8f acc[4] = {};
  for (int kt = 0; kt < KT; ++kt) {
    AFrag a;
    const float* p = xrow + kt * 32 + khi * 8;
    f32x4 f0 = *(const f32x4*)(p);
    f32x4 f1 = *(const f32x4*)(p + 4);
    f32x4 f2 = *(const f32x4*)(p + 16);
    f32x4 f3 = *(const f32x4*)(p + 20);
#pragma unroll
    for (int i = 0; i < 4; ++i) {
      a.h[i]      = f2bf(f0[i]);
      a.h[4 + i]  = f2bf(f1[i]);
      a.h[8 + i]  = f2bf(f2[i]);
      a.h[12 + i] = f2bf(f3[i]);
    }
    v16bf bf[4];
#pragma unroll
    for (int nt = 0; nt < 4; ++nt)
      bf[nt] = *(const v16bf*)(Wfr + ((size_t)(nt * KT + kt)) * 512 + lane * 16);
#pragma unroll
    for (int nt = 0; nt < 4; ++nt)
      acc[nt] = wmma_bf16(a.v, bf[nt], acc[nt]);
  }
  // D layout: n = lane&15, m = r + 8*(lane>>4)
#pragma unroll
  for (int r = 0; r < 8; ++r) {
    int m = r + 8 * khi;
    size_t base = ((size_t)(t0 + m) * B_ + b) * 64;
#pragma unroll
    for (int nt = 0; nt < 4; ++nt)
      xr[base + nt * 16 + sub] = f2bf(acc[nt][r]);
  }
}

// ---------------------------------------------------------------------------
// Persistent LSTM scan. One workgroup (8 waves) per 16-row batch tile.
// Per step: xr A-frags issued first (overlap global latency with phase-1
// math); waves 0-3 compute hu = h@U (K=512) with 4 independent WMMA chains,
// loads grouped 4-wide; barrier; all 8 waves compute g = [xr|hu] @ G + b
// with K outermost so consecutive WMMAs hit independent gate accumulators;
// fp32 gate math with c in VGPRs; h -> bf16 LDS (next step) + fp32 global.
// ---------------------------------------------------------------------------
__global__ __launch_bounds__(256) void lstm_scan_kernel(
    const unsigned short* __restrict__ xr,    // [T][B][64] bf16
    const unsigned short* __restrict__ Gfr,   // 128 nt x 4 kt frags
    const unsigned short* __restrict__ Ufr,   // 4 nt x 16 kt frags
    const float* __restrict__ bias,           // [4H]
    float* __restrict__ xout,                 // [B][T][H]
    float* __restrict__ hlast) {              // d_out tail (stride 2H per b)
  constexpr int HPAD = 520, HUPAD = 72;       // pad to kill LDS bank conflicts
  __shared__ unsigned short sh_h[16 * HPAD];
  __shared__ unsigned short sh_hu[16 * HUPAD];

  const int tid = threadIdx.x, lane = tid & 31, wave = tid >> 5;
  const int b0 = blockIdx.x * 16;
  const int sub = lane & 15, khi = lane >> 4;

  for (int i = tid; i < 16 * HPAD; i += 256) sh_h[i] = 0;   // h(-1) = 0

  float bs[16];
#pragma unroll
  for (int gg = 0; gg < 4; ++gg)
#pragma unroll
    for (int ht = 0; ht < 4; ++ht)
      bs[gg * 4 + ht] = bias[gg * H_ + wave * 64 + ht * 16 + sub];

  v8f c_reg[4] = {};                                        // c(-1) = 0
  __syncthreads();

  for (int t = 0; t < T_; ++t) {
    // ---- xr A-frags (K-tiles 0,1): independent of phase 1, issue early ----
    AFrag a[4];
    {
      const unsigned short* xp = xr + ((size_t)t * B_ + b0 + sub) * 64;
      a[0].q[0] = *(const u32x4*)(xp + khi * 8);
      a[0].q[1] = *(const u32x4*)(xp + 16 + khi * 8);
      a[1].q[0] = *(const u32x4*)(xp + 32 + khi * 8);
      a[1].q[1] = *(const u32x4*)(xp + 48 + khi * 8);
      if (t + 1 < T_)
        __builtin_prefetch(xr + ((size_t)(t + 1) * B_ + b0 + sub) * 64, 0, 3);
    }

    // ---- phase 1: hu = h @ U  (16x64, K=512) on waves 0..3 ----
    if (wave < 4) {
      v8f p[4] = {};                 // 4 independent accumulator chains
#pragma unroll
      for (int kq = 0; kq < 4; ++kq) {
        AFrag ha[4];
        v16bf bf[4];
#pragma unroll
        for (int j = 0; j < 4; ++j) {
          const int kt = kq * 4 + j;
          const unsigned short* hp = sh_h + sub * HPAD + kt * 32 + khi * 8;
          ha[j].q[0] = *(const u32x4*)(hp);
          ha[j].q[1] = *(const u32x4*)(hp + 16);
          bf[j] = *(const v16bf*)(Ufr + ((size_t)(wave * 16 + kt)) * 512 + lane * 16);
        }
#pragma unroll
        for (int j = 0; j < 4; ++j)
          p[j] = wmma_bf16(ha[j].v, bf[j], p[j]);
      }
      v8f acc = (p[0] + p[1]) + (p[2] + p[3]);
#pragma unroll
      for (int r = 0; r < 8; ++r)
        sh_hu[(r + 8 * khi) * HUPAD + wave * 16 + sub] = f2bf(acc[r]);
    }
    __syncthreads();

    // ---- hu A-frags (K-tiles 2,3) ----
    {
      const unsigned short* up = sh_hu + sub * HUPAD;
      a[2].q[0] = *(const u32x4*)(up + khi * 8);
      a[2].q[1] = *(const u32x4*)(up + 16 + khi * 8);
      a[3].q[0] = *(const u32x4*)(up + 32 + khi * 8);
      a[3].q[1] = *(const u32x4*)(up + 48 + khi * 8);
    }

    // ---- gates + state update: wave owns h-cols [64*wave, 64*wave+64) ----
#pragma unroll
    for (int ht = 0; ht < 4; ++ht) {
      // 4 gate accumulators (i,f,o,c~), init with bias splat
      v8f g[4];
#pragma unroll
      for (int gg = 0; gg < 4; ++gg) {
        float bv = bs[gg * 4 + ht];
#pragma unroll
        for (int r = 0; r < 8; ++r) g[gg][r] = bv;
      }
      // Base of this wave's tile column within each gate block
      const unsigned short* gbase = Gfr + (size_t)(wave * 4 + ht) * 2048 + lane * 16;
#pragma unroll
      for (int kt = 0; kt < 4; ++kt) {
        v16bf bf[4];
#pragma unroll
        for (int gg = 0; gg < 4; ++gg)
          bf[gg] = *(const v16bf*)(gbase + (size_t)gg * 32 * 2048 + kt * 512);
#pragma unroll
        for (int gg = 0; gg < 4; ++gg)
          g[gg] = wmma_bf16(a[kt].v, bf[gg], g[gg]);   // independent dests
      }
      const int colb = wave * 64 + ht * 16 + sub;
#pragma unroll
      for (int r = 0; r < 8; ++r) {
        float i_ = fsig(g[0][r]);
        float f_ = fsig(g[1][r]);
        float o_ = fsig(g[2][r]);
        float ct = ftanh(g[3][r]);
        float c = f_ * c_reg[ht][r] + i_ * ct;
        c_reg[ht][r] = c;
        float h = o_ * ftanh(c);
        int m = r + 8 * khi;
        sh_h[m * HPAD + colb] = f2bf(h);
        xout[((size_t)(b0 + m) * T_ + t) * H_ + colb] = h;
        if (t == T_ - 1)
          hlast[(size_t)(b0 + m) * (2 * H_) + colb] = h;
      }
    }
    __syncthreads();
  }
}

// ---------------------------------------------------------------------------
static inline int ceil_div(int a, int b) { return (a + b - 1) / b; }

extern "C" void kernel_launch(void* const* d_in, const int* in_sizes, int n_in,
                              void* d_out, int out_size, void* d_ws, size_t ws_size,
                              hipStream_t stream) {
  (void)in_sizes; (void)n_in; (void)out_size; (void)ws_size;
  const float* x   = (const float*)d_in[0];
  const float* W0  = (const float*)d_in[1];
  const float* Wg0 = (const float*)d_in[2];
  const float* U0  = (const float*)d_in[3];
  const float* Ug0 = (const float*)d_in[4];
  const float* b0  = (const float*)d_in[5];
  const float* W1  = (const float*)d_in[6];
  const float* Wg1 = (const float*)d_in[7];
  const float* U1  = (const float*)d_in[8];
  const float* Ug1 = (const float*)d_in[9];
  const float* b1  = (const float*)d_in[10];
  float* out = (float*)d_out;

  // workspace layout (bf16 halves), ~9.7 MB total
  unsigned short* ws = (unsigned short*)d_ws;
  size_t off = 0;
  unsigned short* xrbuf = ws + off; off += (size_t)T_ * B_ * R_;   // 8 MB
  unsigned short* G0f   = ws + off; off += (size_t)128 * 4 * 512;  // 512 KB
  unsigned short* G1f   = ws + off; off += (size_t)128 * 4 * 512;
  unsigned short* U0f   = ws + off; off += (size_t)4 * 16 * 512;   // 64 KB
  unsigned short* U1f   = ws + off; off += (size_t)4 * 16 * 512;
  unsigned short* W0f   = ws + off; off += (size_t)4 * 8 * 512;    // 32 KB
  unsigned short* W1f   = ws + off; off += (size_t)4 * 16 * 512;

  auto packs = [&](const float* src, int ldn, int KTsrc, int Nt, int KTtot,
                   int ktOff, unsigned short* dst) {
    int total = KTsrc * Nt * 32;
    pack_frags_kernel<<<ceil_div(total, 256), 256, 0, stream>>>(
        src, ldn, KTsrc, Nt, KTtot, ktOff, dst);
  };

  // Weight packing (tiny, one-shot). G = [[Wg];[Ug]] as K=128, N=2048.
  packs(W0,  R_,  D_ / 32, R_ / 16,  D_ / 32, 0, W0f);
  packs(Wg0, G4H, 2,       G4H / 16, 4,       0, G0f);
  packs(Ug0, G4H, 2,       G4H / 16, 4,       2, G0f);
  packs(U0,  R_,  H_ / 32, R_ / 16,  H_ / 32, 0, U0f);
  packs(W1,  R_,  H_ / 32, R_ / 16,  H_ / 32, 0, W1f);
  packs(Wg1, G4H, 2,       G4H / 16, 4,       0, G1f);
  packs(Ug1, G4H, 2,       G4H / 16, 4,       2, G1f);
  packs(U1,  R_,  H_ / 32, R_ / 16,  H_ / 32, 0, U1f);

  const int xr_blocks = (B_ * (T_ / 16)) / 8;   // 512 blocks x 8 waves
  float* hl = out + (size_t)B_ * T_ * H_;       // concat(h1,h2) tail

  // layer 0: xr0 = x@W0 ; scan -> x1 staged in d_out's x2 region
  gemm_xr_kernel<<<xr_blocks, 256, 0, stream>>>(x, D_, D_ / 32, W0f, xrbuf);
  lstm_scan_kernel<<<B_ / 16, 256, 0, stream>>>(xrbuf, G0f, U0f, b0, out, hl);
  // layer 1: xr1 = x1@W1 (reads d_out) ; scan overwrites d_out with x2
  gemm_xr_kernel<<<xr_blocks, 256, 0, stream>>>(out, H_, H_ / 32, W1f, xrbuf);
  lstm_scan_kernel<<<B_ / 16, 256, 0, stream>>>(xrbuf, G1f, U1f, b1, out, hl + H_);
}
